// SentenceEncoder_25881472926222
// MI455X (gfx1250) — matrix operand order repaired and checked by reading, not compiled, not run on hardware
//
#include <hip/hip_runtime.h>
#include <hip/hip_bf16.h>

typedef __attribute__((ext_vector_type(16))) __bf16 v16bf;
typedef __attribute__((ext_vector_type(8)))  float  v8f;

#define L_DIM 128
#define B_DIM 2048
#define D_DIM 300
#define H_DIM 60
#define KPAD  320   // D padded to multiple of 32
#define NPAD  64    // H padded to multiple of 16

// ---------------------------------------------------------------------------
// Kernel 1: projected = normalize_by_sqnorm( sentence[L*B, D] x proj[D, H] )
// bf16 WMMA 16x16x32, f32 accumulate. One wave = 16 rows x 64 cols tile.
// Block = 256 threads = 8 waves = 128 rows. Grid = (L*B)/128 = 2048 blocks.
// K-loop is peeled (8 pipelined steps + step 8 + guarded tail) so the hot
// loop has no data-dependent branches and A-tile global loads overlap WMMA.
// ---------------------------------------------------------------------------
__global__ __launch_bounds__(256) void proj_gemm_kernel(
    const float* __restrict__ sentence,   // [L*B, D]
    const float* __restrict__ proj,       // [D, H]
    float* __restrict__ projn)            // [L*B, H] normalized output
{
  // proj, transposed to [n][k], zero padded, bf16: 64*320*2 = 40KB LDS
  __shared__ __bf16 projT[NPAD * KPAD];

  const int t = threadIdx.x;
  for (int idx = t; idx < NPAD * KPAD; idx += 256) {
    int n = idx / KPAD;
    int k = idx - n * KPAD;
    float v = (n < H_DIM && k < D_DIM) ? proj[k * H_DIM + n] : 0.0f;
    projT[idx] = (__bf16)v;
  }
  __syncthreads();

  const int wave = t >> 5;
  const int lane = t & 31;
  const int g    = lane >> 4;   // half-wave group (0/1)
  const int m    = lane & 15;   // row within tile (A) / col within n-tile (B,C)

  const long rowBase = ((long)blockIdx.x * 8 + wave) * 16;
  const float* __restrict__ arow = sentence + (rowBase + m) * (long)D_DIM;

  v8f acc[4];
#pragma unroll
  for (int nt = 0; nt < 4; ++nt) acc[nt] = (v8f)0.0f;

  // raw A staging registers (float4 x4 = one 16x32 fragment's worth per lane)
  float4 x0, x1, x2, x3;
  auto loadA = [&](int k0) {
    const float4* p0 = (const float4*)(arow + k0 + 8 * g);
    const float4* p1 = (const float4*)(arow + k0 + 16 + 8 * g);
    x0 = p0[0]; x1 = p0[1]; x2 = p1[0]; x3 = p1[1];
  };
  auto cvtA = [&]() {
    // 16-bit A 16x32 layout: lane group g, element e -> K = k0 + (e/8)*16 + 8g + (e%8)
    v16bf a;
    a[0]  = (__bf16)x0.x; a[1]  = (__bf16)x0.y; a[2]  = (__bf16)x0.z; a[3]  = (__bf16)x0.w;
    a[4]  = (__bf16)x1.x; a[5]  = (__bf16)x1.y; a[6]  = (__bf16)x1.z; a[7]  = (__bf16)x1.w;
    a[8]  = (__bf16)x2.x; a[9]  = (__bf16)x2.y; a[10] = (__bf16)x2.z; a[11] = (__bf16)x2.w;
    a[12] = (__bf16)x3.x; a[13] = (__bf16)x3.y; a[14] = (__bf16)x3.z; a[15] = (__bf16)x3.w;
    return a;
  };
  auto mmac4 = [&](v16bf a, int k0) {
    // B fragment from LDS: lane group g, element e -> K = k0 + (e/8)*16 + 8*g + (e%8)
#pragma unroll
    for (int nt = 0; nt < 4; ++nt) {
      const int n = nt * 16 + m;
      const __bf16* pb = &projT[n * KPAD + k0 + 8 * g];
      v16bf b;
#pragma unroll
      for (int e = 0; e < 8; ++e) { b[e] = pb[e]; b[e + 8] = pb[16 + e]; }
      acc[nt] = __builtin_amdgcn_wmma_f32_16x16x32_bf16(
          false, a, false, b, (short)0, acc[nt], false, false);
    }
  };

  loadA(0);
#pragma unroll 1
  for (int ks = 0; ks < 8; ++ks) {
    v16bf a = cvtA();
    loadA((ks + 1) * 32);     // prefetch next step (max col 287 < 300) before WMMAs
    mmac4(a, ks * 32);
  }
  // step 8 (k0 = 256): uses data prefetched by last loop iteration
  mmac4(cvtA(), 256);
  // tail step (k0 = 288): cols 288+8g..295+8g valid only below 300, rest zero
  {
    v16bf a;
#pragma unroll
    for (int e = 0; e < 16; ++e) a[e] = (__bf16)0.0f;
#pragma unroll
    for (int e = 0; e < 8; ++e) {
      int col = 288 + 8 * g + e;
      if (col < D_DIM) a[e] = (__bf16)arow[col];
    }
    mmac4(a, 288);
  }

  // ---- fused normalization: per output row, inv = 1/max(sum_h p^2, 1e-5)
  // C layout: VGPR r holds row (r + 8*g), col = nt*16 + m. Reduce over the
  // 16 lanes of each half-wave (masks 1,2,4,8 stay inside the half).
  float inv[8];
#pragma unroll
  for (int r = 0; r < 8; ++r) {
    float s = acc[0][r] * acc[0][r] + acc[1][r] * acc[1][r] +
              acc[2][r] * acc[2][r] + acc[3][r] * acc[3][r];
    s += __shfl_xor(s, 1, 32);
    s += __shfl_xor(s, 2, 32);
    s += __shfl_xor(s, 4, 32);
    s += __shfl_xor(s, 8, 32);
    inv[r] = 1.0f / fmaxf(s, 1e-5f);
  }

#pragma unroll
  for (int r = 0; r < 8; ++r) {
    const long row = rowBase + r + 8 * g;
#pragma unroll
    for (int nt = 0; nt < 4; ++nt) {
      const int n = nt * 16 + m;
      if (n < H_DIM) projn[row * H_DIM + n] = acc[nt][r] * inv[r];
    }
  }
}

// ---------------------------------------------------------------------------
// Kernel 2: sim[l,b] = dot(projn[l,b,:], projn[l-1,b,:]); sim[0,b] = -inf
// One wave per b, walks l keeping previous row in registers. projn is
// L2-resident (63MB < 192MB), so these reads mostly hit L2.
// ---------------------------------------------------------------------------
__global__ __launch_bounds__(256) void sim_kernel(
    const float* __restrict__ projn, float* __restrict__ sim)
{
  const int wave = threadIdx.x >> 5;
  const int lane = threadIdx.x & 31;
  const int b = blockIdx.x * 8 + wave;
  const int h = lane * 2;

  float2 prev = {0.0f, 0.0f};
#pragma unroll 1
  for (int l = 0; l < L_DIM; ++l) {
    float2 cur = {0.0f, 0.0f};
    if (h < H_DIM)
      cur = *(const float2*)(projn + ((long)l * B_DIM + b) * H_DIM + h);
    float part = cur.x * prev.x + cur.y * prev.y;
    part += __shfl_xor(part, 16, 32);
    part += __shfl_xor(part, 8, 32);
    part += __shfl_xor(part, 4, 32);
    part += __shfl_xor(part, 2, 32);
    part += __shfl_xor(part, 1, 32);
    if (lane == 0)
      sim[(long)l * B_DIM + b] = (l == 0) ? -__builtin_inff() : part;
    prev = cur;
  }
}

// ---------------------------------------------------------------------------
// Kernel 3: masked 3-way softmax -> packed weights float4{w0,w1,w2,0} [L,B]
// ---------------------------------------------------------------------------
__global__ __launch_bounds__(256) void weights_kernel(
    const float* __restrict__ sim, const int* __restrict__ size,
    float4* __restrict__ W)
{
  const int idx = blockIdx.x * 256 + threadIdx.x;
  if (idx >= L_DIM * B_DIM) return;
  const int l = idx / B_DIM;
  const int b = idx - l * B_DIM;

  const float NEG = -__builtin_inff();
  float s0 = sim[idx];                                   // sim[0,:] == -inf
  float s2 = (l < L_DIM - 1) ? sim[idx + B_DIM] : NEG;
  const int sz = size[b];
  float w0 = (l < sz) ? s0 : NEG;
  const int sz2 = (sz - 1 > 0) ? sz - 1 : 0;
  float w2 = (l < sz2) ? s2 : NEG;

  float mx = fmaxf(1.0f, fmaxf(w0, w2));
  float e0 = __expf(w0 - mx);                            // exp(-inf) -> 0
  float e1 = __expf(1.0f - mx);
  float e2 = __expf(w2 - mx);
  float inv = 1.0f / (e0 + e1 + e2);
  float4 w = {e0 * inv, e1 * inv, e2 * inv, 0.0f};
  W[idx] = w;
}

// ---------------------------------------------------------------------------
// Kernel 4: out[l,b,:] = w1*s[l] + w0*s[l-1] + w2*s[l+1]
// One thread per (b,d); prev/cur/next kept in registers -> sentence read once.
// sentence/out are 315MB pure streams: non-temporal so they don't evict the
// L2-resident 4MB weight plane (one b128 broadcast load per l per wave).
// ---------------------------------------------------------------------------
__global__ __launch_bounds__(256) void combine_kernel(
    const float* __restrict__ sentence,
    const float4* __restrict__ W,
    float* __restrict__ out)
{
  const long tid = (long)blockIdx.x * 256 + threadIdx.x;
  if (tid >= (long)B_DIM * D_DIM) return;
  const int b = (int)(tid / D_DIM);
  const long stride = (long)B_DIM * D_DIM;
  const long base = tid;  // b*D + d

  float prev = 0.0f;
  float cur  = __builtin_nontemporal_load(sentence + base);
#pragma unroll 1
  for (int l = 0; l < L_DIM; ++l) {
    float nxt = (l < L_DIM - 1)
                    ? __builtin_nontemporal_load(sentence + base + (long)(l + 1) * stride)
                    : 0.0f;
    const float4 w = W[(long)l * B_DIM + b];
    float o = w.y * cur + w.x * prev + w.z * nxt;
    __builtin_nontemporal_store(o, out + base + (long)l * stride);
    prev = cur;
    cur = nxt;
  }
}

// ---------------------------------------------------------------------------
extern "C" void kernel_launch(void* const* d_in, const int* in_sizes, int n_in,
                              void* d_out, int out_size, void* d_ws, size_t ws_size,
                              hipStream_t stream) {
  const float* sentence = (const float*)d_in[0];   // [L,B,D] f32
  const int*   size     = (const int*)d_in[1];     // [B] i32
  const float* proj     = (const float*)d_in[2];   // [D,H] f32
  float* out = (float*)d_out;

  float* ws    = (float*)d_ws;
  float* projn = ws;                                   // L*B*H floats (~63MB)
  float* sim   = projn + (long)L_DIM * B_DIM * H_DIM;  // L*B floats
  float4* W    = (float4*)(sim + (long)L_DIM * B_DIM); // L*B float4 (16B-aligned)

  proj_gemm_kernel<<<(L_DIM * B_DIM) / 128, 256, 0, stream>>>(sentence, proj, projn);
  sim_kernel<<<B_DIM / 8, 256, 0, stream>>>(projn, sim);
  weights_kernel<<<(L_DIM * B_DIM + 255) / 256, 256, 0, stream>>>(sim, size, W);
  combine_kernel<<<(int)(((long)B_DIM * D_DIM + 255) / 256), 256, 0, stream>>>(
      sentence, W, out);
}
